// WAHead_85152021610607
// MI455X (gfx1250) — matrix operand order repaired and checked by reading, not compiled, not run on hardware
//
#include <hip/hip_runtime.h>
#include <hip/hip_bf16.h>

// Problem constants (from reference)
#define BSZ   32768
#define KK    5
#define EE    16
#define LL    128
#define HH    3
#define NROWS (BSZ * KK)           // 163840
#define ROWS_PER_BLK 32
#define NBLK  (NROWS / ROWS_PER_BLK) // 5120

typedef __attribute__((ext_vector_type(16))) _Float16 v16h;
typedef __attribute__((ext_vector_type(8)))  _Float16 v8h;
typedef __attribute__((ext_vector_type(8)))  float    v8f;

// Branch-free fast activations: v_exp_f32 + v_rcp_f32, no libm range-reduction.
__device__ __forceinline__ float fast_rcp(float x) { return __builtin_amdgcn_rcpf(x); }
__device__ __forceinline__ float sigf(float x)     { return fast_rcp(1.0f + __expf(-x)); }
// tanh(x) = 1 - 2/(exp(2x)+1); saturates to +/-1 correctly as exp -> inf/0.
__device__ __forceinline__ float tanhfast(float x) { return fmaf(-2.0f, fast_rcp(1.0f + __expf(2.0f * x)), 1.0f); }

__device__ __forceinline__ v16h pack16(v8h lo, v8h hi8) {
    v16h a;
#pragma unroll
    for (int q = 0; q < 8; ++q) { a[q] = lo[q]; a[q + 8] = hi8[q]; }
    return a;
}

// ---------------------------------------------------------------------------
// Prep: convert W_ih (512x16) and W_hh (512x128) to f16, sum biases.
// ---------------------------------------------------------------------------
__global__ void prep_kernel(const float* __restrict__ wih, const float* __restrict__ whh,
                            const float* __restrict__ bih, const float* __restrict__ bhh,
                            _Float16* __restrict__ wih16, _Float16* __restrict__ whh16,
                            float* __restrict__ bias) {
    int i = blockIdx.x * 256 + threadIdx.x;
    if (i < 512 * 128) whh16[i] = (_Float16)whh[i];
    if (i < 512 * 16)  wih16[i] = (_Float16)wih[i];
    if (i < 512)       bias[i]  = bih[i] + bhh[i];
}

// ---------------------------------------------------------------------------
// Main fused kernel: anchor MLP + 3-step LSTM (WMMA f16->f32) + heads.
// One block = 32 rows (two 16-row groups). Wave w owns hidden slice
// l in [16w,16w+16) and its i/f/g/o gate tiles -> LSTM update in registers.
// ---------------------------------------------------------------------------
__global__ __launch_bounds__(256) void lstm_head_kernel(
    const float* __restrict__ z, const float* __restrict__ anchors,
    const float* __restrict__ W1, const float* __restrict__ b1,
    const float* __restrict__ W2, const float* __restrict__ b2,
    const _Float16* __restrict__ wih16, const _Float16* __restrict__ whh16,
    const float* __restrict__ bias,
    const float* __restrict__ Wp, const float* __restrict__ bp,
    const float* __restrict__ Wr, const float* __restrict__ br,
    float* __restrict__ out)
{
    __shared__ __align__(16) _Float16 h16[2][16 * 128];   // per-rowgroup h (f16), 8KB
    __shared__ __align__(16) float    h32[32 * 128];      // final-step h (f32), 16KB
    __shared__ __align__(16) _Float16 emb16[2][16 * 32];  // emb padded K=16->32, 2KB
    __shared__ float risk_s[32][3];

    const int tid  = threadIdx.x;
    const int lane = tid & 31;
    const int wave = tid >> 5;          // 0..7
    const int n    = lane & 15;         // tile row (A) / tile col (B,C,D)
    const int hi   = lane >> 4;         // half-wave select
    const int rowbase = blockIdx.x * ROWS_PER_BLK;

    // ---- init h = z (broadcast over K anchors) ----
#pragma unroll
    for (int it = 0; it < 16; ++it) {
        int idx = tid + it * 256;       // 0..4095
        int m = idx >> 7, l = idx & 127;
        int row = rowbase + m;
        float zv = z[(row / KK) * LL + l];
        h16[m >> 4][(m & 15) * 128 + l] = (_Float16)zv;
    }
    // ---- anchor embedding MLP: emb = relu(a@W1.T+b1)@W2.T+b2 ----
#pragma unroll
    for (int it = 0; it < 2; ++it) {
        int idx = tid + it * 256;       // 0..511
        int m = idx >> 4, j = idx & 15;
        int row = rowbase + m;
        float a0 = anchors[row * 2 + 0], a1 = anchors[row * 2 + 1];
        float s = b2[j];
#pragma unroll
        for (int mm = 0; mm < 16; ++mm) {
            float ha = fmaf(W1[mm * 2], a0, fmaf(W1[mm * 2 + 1], a1, b1[mm]));
            ha = fmaxf(ha, 0.0f);
            s = fmaf(W2[j * 16 + mm], ha, s);
        }
        emb16[m >> 4][(m & 15) * 32 + j]      = (_Float16)s;
        emb16[m >> 4][(m & 15) * 32 + 16 + j] = (_Float16)0.0f;  // zero K-pad
    }

    const int colbase = 16 * wave;  // this wave's l-slice base
    float biasv[4];
#pragma unroll
    for (int g = 0; g < 4; ++g) biasv[g] = bias[g * 128 + colbase + n];

    // ---- B fragments of W_hh.T, cached in registers for all steps ----
    // B 32x16 f16 layout: lane n = column, halves[0..7] = K = kc*32 + hi*8 + (0..7), [8..15] = K = kc*32 + 16 + hi*8 + (0..7)
    // B[k][col] = W_hh[colglob][k] -> two 8-half runs of one W_hh row.
    v16h bfrag[4][4];
#pragma unroll
    for (int g = 0; g < 4; ++g) {
        const _Float16* wrow = whh16 + (g * 128 + colbase + n) * 128;
#pragma unroll
        for (int kc = 0; kc < 4; ++kc) {
            v8h l0 = *(const v8h*)(wrow + kc * 32 + hi * 8);
            v8h l1 = *(const v8h*)(wrow + kc * 32 + 16 + hi * 8);
            bfrag[g][kc] = pack16(l0, l1);
        }
    }
    // ---- W_ih B fragments (K=16 real, 16..31 zero => lanes 16..31 zero) ----
    v16h bwih[4];
#pragma unroll
    for (int g = 0; g < 4; ++g) {
        v16h b;
#pragma unroll
        for (int q = 0; q < 16; ++q) b[q] = (_Float16)0.0f;
        if (hi == 0) {
            const _Float16* wrow = wih16 + (g * 128 + colbase + n) * 16;
            v8h l0 = *(const v8h*)(wrow);
            v8h l1 = *(const v8h*)(wrow + 8);
            b = pack16(l0, l1);
        }
        bwih[g] = b;
    }

    __syncthreads();  // emb16 / h16 init visible

    // ---- A fragments of padded emb (constant across steps) ----
    // A 16x32 f16, this code's contiguous k order (as bwih): lane n = row, halves[0..15]=K=hi*16+(0..15)
    v16h aemb[2];
#pragma unroll
    for (int rg = 0; rg < 2; ++rg) {
        const _Float16* erow = emb16[rg] + n * 32;
        v8h l0 = *(const v8h*)(erow + hi * 16);
        v8h l1 = *(const v8h*)(erow + hi * 16 + 8);
        aemb[rg] = pack16(l0, l1);
    }

    // ---- LSTM cell state in registers: lane holds c for 8 rows at its l ----
    v8f creg[2];
#pragma unroll
    for (int rg = 0; rg < 2; ++rg)
#pragma unroll
        for (int r = 0; r < 8; ++r) creg[rg][r] = 0.0f;

    // ================= 3-step rollout =================
#pragma unroll
    for (int step = 0; step < HH; ++step) {
#pragma unroll
        for (int rg = 0; rg < 2; ++rg) {
            __syncthreads();  // prior writes to h16[rg] complete
            v16h afrag[4];
#pragma unroll
            for (int kc = 0; kc < 4; ++kc) {
                const _Float16* hrow = h16[rg] + n * 128 + kc * 32;
                v8h l0 = *(const v8h*)(hrow + hi * 8);
                v8h l1 = *(const v8h*)(hrow + 16 + hi * 8);
                afrag[kc] = pack16(l0, l1);
            }
            __syncthreads();  // all reads of h16[rg] done before rewrite

            // gates = bias + h@W_hh.T + emb@W_ih.T  (5 chained WMMAs / gate)
            v8f acc[4];
#pragma unroll
            for (int g = 0; g < 4; ++g) {
                v8f c;
#pragma unroll
                for (int r = 0; r < 8; ++r) c[r] = biasv[g];
#pragma unroll
                for (int kc = 0; kc < 4; ++kc)
                    c = __builtin_amdgcn_wmma_f32_16x16x32_f16(
                            false, afrag[kc], false, bfrag[g][kc], (short)0, c, false, false);
                c = __builtin_amdgcn_wmma_f32_16x16x32_f16(
                        false, aemb[rg], false, bwih[g], (short)0, c, false, false);
                acc[g] = c;
            }

            // register-resident LSTM update; C layout: row m = r + 8*hi, col = colbase+n
#pragma unroll
            for (int r = 0; r < 8; ++r) {
                float ig = sigf(acc[0][r]);
                float fg = sigf(acc[1][r]);
                float gg = tanhfast(acc[2][r]);
                float og = sigf(acc[3][r]);
                float cv = fmaf(fg, creg[rg][r], ig * gg);
                creg[rg][r] = cv;
                float hv = og * tanhfast(cv);
                int mloc = r + 8 * hi;
                int l = colbase + n;
                h16[rg][mloc * 128 + l] = (_Float16)hv;
                if (step == HH - 1) h32[(rg * 16 + mloc) * 128 + l] = hv;
            }
        }
    }
    __syncthreads();

    // ================= heads =================
    if (tid < 128) {
        int m = tid >> 2, j = tid & 3;
        const float* hv = &h32[m * 128];
        int row = rowbase + m;
        if (j == 3) {
            float s = bp[0];
#pragma unroll 16
            for (int l = 0; l < 128; ++l) s = fmaf(Wp[l], hv[l], s);
            out[NROWS + row] = s;                       // progress
        } else {
            float s = br[j];
#pragma unroll 16
            for (int l = 0; l < 128; ++l) s = fmaf(Wr[j * 128 + l], hv[l], s);
            risk_s[m][j] = s;
            out[3 * NROWS + j * NROWS + row] = s;       // risks[e]
        }
    }
    __syncthreads();
    if (tid < 32) {
        int m = tid, row = rowbase + m;
        float r0 = risk_s[m][0], r1 = risk_s[m][1], r2 = risk_s[m][2];
        out[row] = (r0 + r1 + r2) * (1.0f / 3.0f);      // risk_mean
        float s0 = sigf(r0), s1 = sigf(r1), s2 = sigf(r2);
        float sm = (s0 + s1 + s2) * (1.0f / 3.0f);
        float d0 = s0 - sm, d1 = s1 - sm, d2 = s2 - sm;
        out[2 * NROWS + row] = (d0 * d0 + d1 * d1 + d2 * d2) * 0.5f;  // unbiased var
    }
}

extern "C" void kernel_launch(void* const* d_in, const int* in_sizes, int n_in,
                              void* d_out, int out_size, void* d_ws, size_t ws_size,
                              hipStream_t stream) {
    const float* z    = (const float*)d_in[0];
    const float* anch = (const float*)d_in[1];
    const float* W1   = (const float*)d_in[2];
    const float* b1   = (const float*)d_in[3];
    const float* W2   = (const float*)d_in[4];
    const float* b2   = (const float*)d_in[5];
    const float* Wih  = (const float*)d_in[6];
    const float* Whh  = (const float*)d_in[7];
    const float* bih  = (const float*)d_in[8];
    const float* bhh  = (const float*)d_in[9];
    const float* Wp   = (const float*)d_in[10];
    const float* bp   = (const float*)d_in[11];
    const float* Wr   = (const float*)d_in[12];
    const float* br   = (const float*)d_in[13];

    // Workspace: W_hh f16 (128KB) | W_ih f16 (16KB) | bias sum (2KB)
    _Float16* whh16 = (_Float16*)d_ws;
    _Float16* wih16 = (_Float16*)((char*)d_ws + 512 * 128 * 2);
    float*    bias  = (float*)((char*)d_ws + 512 * 128 * 2 + 512 * 16 * 2);

    prep_kernel<<<256, 256, 0, stream>>>(Wih, Whh, bih, bhh, wih16, whh16, bias);
    lstm_head_kernel<<<NBLK, 256, 0, stream>>>(z, anch, W1, b1, W2, b2,
                                               wih16, whh16, bias,
                                               Wp, bp, Wr, br, (float*)d_out);
}
